// PointNetSetAbstraction_3152505995469
// MI455X (gfx1250) — compile-verified, matrix operand order checked by reading, not run on hardware
//
#include <hip/hip_runtime.h>
#include <hip/hip_bf16.h>

// ---------------------------------------------------------------------------
// PointNet++ Set Abstraction for MI455X (gfx1250, wave32, WMMA)
// v2: channel-contiguous [M][C] activation layout -> b128 loads/stores in the
//     WMMA GEMM, BN statistics fused into the GEMM (shfl + LDS atomics),
//     wave-cooperative NS-maxpool. Conv bias absorbed by BatchNorm (skipped).
// ---------------------------------------------------------------------------

#define B_  16
#define N_  4096
#define S_  512
#define NS_ 32
#define MTOT (B_ * S_ * NS_)          // 262144 columns
#define RAD2 0.16f                    // 0.4^2
#define EPS_ 1e-05f

typedef __attribute__((ext_vector_type(16))) _Float16 v16h;
typedef __attribute__((ext_vector_type(8)))  _Float16 v8h;   // 16 bytes
typedef __attribute__((ext_vector_type(8)))  float    v8f;

// ---------------------------------------------------------------------------
// 1) Farthest-point sampling: one workgroup per batch, sequential 512 steps.
// ---------------------------------------------------------------------------
__global__ void fps_kernel(const float* __restrict__ xyz,   // (B,3,N)
                           int* __restrict__ fps_idx,       // (B,S)
                           float* __restrict__ new_xyz) {   // (B,S,3)
    const int b = blockIdx.x;
    const int t = threadIdx.x;                 // 256 threads
    const float* xb = xyz + (size_t)b * 3 * N_;

    __shared__ float s_val[256];
    __shared__ int   s_idx[256];
    __shared__ float s_cent[3];

    float dist[16];
#pragma unroll
    for (int k = 0; k < 16; ++k) dist[k] = 1e10f;

    int far = 0;
    for (int s = 0; s < S_; ++s) {
        if (t == 0) {
            fps_idx[b * S_ + s] = far;
            float cx = xb[far], cy = xb[N_ + far], cz = xb[2 * N_ + far];
            s_cent[0] = cx; s_cent[1] = cy; s_cent[2] = cz;
            float* o = new_xyz + ((size_t)b * S_ + s) * 3;
            o[0] = cx; o[1] = cy; o[2] = cz;
        }
        __syncthreads();
        const float cx = s_cent[0], cy = s_cent[1], cz = s_cent[2];

        float best = -1.0f; int bi = 0;
#pragma unroll
        for (int k = 0; k < 16; ++k) {
            const int n = t + k * 256;
            const float dx = xb[n] - cx, dy = xb[N_ + n] - cy, dz = xb[2 * N_ + n] - cz;
            const float d  = dx * dx + dy * dy + dz * dz;
            const float nd = fminf(dist[k], d);
            dist[k] = nd;
            if (nd > best) { best = nd; bi = n; }
        }
        s_val[t] = best; s_idx[t] = bi;
        __syncthreads();
        for (int off = 128; off > 0; off >>= 1) {
            if (t < off) {
                const float ov = s_val[t + off]; const int oi = s_idx[t + off];
                if (ov > s_val[t] || (ov == s_val[t] && oi < s_idx[t])) {
                    s_val[t] = ov; s_idx[t] = oi;
                }
            }
            __syncthreads();
        }
        far = s_idx[0];
        __syncthreads();
    }
}

// ---------------------------------------------------------------------------
// 2) Ball query: one wave per (b,s); ballot/prefix append = first NS sorted
//    in-radius indices, remainder filled with the first hit.
// ---------------------------------------------------------------------------
__global__ void query_ball_kernel(const float* __restrict__ xyz,     // (B,3,N)
                                  const float* __restrict__ new_xyz, // (B,S,3)
                                  int* __restrict__ grp) {           // (B,S,NS)
    const int wid  = (int)((blockIdx.x * blockDim.x + threadIdx.x) >> 5);
    const int lane = threadIdx.x & 31;
    if (wid >= B_ * S_) return;
    const int b = wid / S_;
    const float* xb = xyz + (size_t)b * 3 * N_;
    const float cx = new_xyz[wid * 3 + 0];
    const float cy = new_xyz[wid * 3 + 1];
    const float cz = new_xyz[wid * 3 + 2];
    int* g = grp + (size_t)wid * NS_;

    int cnt = 0, first = 0;
    for (int base = 0; base < N_ && cnt < NS_; base += 32) {
        const int n = base + lane;
        const float dx = xb[n] - cx, dy = xb[N_ + n] - cy, dz = xb[2 * N_ + n] - cz;
        const bool pred = (dx * dx + dy * dy + dz * dz) <= RAD2;
        const unsigned mask = (unsigned)__ballot(pred);
        if (cnt == 0 && mask != 0u) first = base + __ffs(mask) - 1;
        const int pos = cnt + __popc(mask & ((1u << lane) - 1u));
        if (pred && pos < NS_) g[pos] = n;
        cnt += __popc(mask);
    }
    if (cnt < NS_ && lane >= cnt) g[lane] = first;
}

// ---------------------------------------------------------------------------
// 3) Feature build: X0[f16] layout [M][8] (6 channels + 2 zero pad), one
//    16-byte store per column. Column m = (b*S+s)*NS + ns.
// ---------------------------------------------------------------------------
__global__ void build_feat_kernel(const float* __restrict__ xyz,
                                  const float* __restrict__ points,
                                  const float* __restrict__ new_xyz,
                                  const int* __restrict__ grp,
                                  _Float16* __restrict__ X0) {
    const int m = blockIdx.x * blockDim.x + threadIdx.x;
    if (m >= MTOT) return;
    const int bs = m / NS_;
    const int b  = bs / S_;
    const int idx = grp[m];
    const float* xb = xyz    + (size_t)b * 3 * N_;
    const float* pb = points + (size_t)b * 3 * N_;
    v8h v;
    v[0] = (_Float16)(xb[idx]          - new_xyz[bs * 3 + 0]);
    v[1] = (_Float16)(xb[N_ + idx]     - new_xyz[bs * 3 + 1]);
    v[2] = (_Float16)(xb[2 * N_ + idx] - new_xyz[bs * 3 + 2]);
    v[3] = (_Float16)pb[idx];
    v[4] = (_Float16)pb[N_ + idx];
    v[5] = (_Float16)pb[2 * N_ + idx];
    v[6] = (_Float16)0.0f;
    v[7] = (_Float16)0.0f;
    *(v8h*)(X0 + (size_t)m * 8) = v;
}

// ---------------------------------------------------------------------------
// 4) WMMA GEMM with fused BN statistics.
//    X layout [M][CSTORE] f16, Y layout [M][cout] f16 (cout = gridDim.y*16).
//    One wave per 16x16 tile. CDNA5 16-bit fragment packing (CDNA5 ISA):
//      lane<16: halves 0..7 = K ks+0..7,  8..15 = K ks+16..23 (row/col = lane)
//      lane>=16: halves 0..7 = K ks+8..15, 8..15 = K ks+24..31
//    -> with [M][C] each half-chunk is one aligned 16-byte load/store.
//    BN stats: shfl_xor column-sum -> LDS ds_add_f32 -> global atomicAdd.
// ---------------------------------------------------------------------------
template <int KTOT, int CSTORE, bool AFFINE>
__global__ void gemm_wmma_kernel(const float* __restrict__ W, int cin,
                                 const _Float16* __restrict__ X,
                                 const float* __restrict__ aff_a,
                                 const float* __restrict__ aff_c,
                                 _Float16* __restrict__ Y,
                                 float* __restrict__ sum,
                                 float* __restrict__ sumsq) {
    const int wave = threadIdx.x >> 5;
    const int lane = threadIdx.x & 31;
    const int mt   = blockIdx.x * (blockDim.x >> 5) + wave;  // column tile
    const int ct   = blockIdx.y;                             // row tile
    const int cout = (int)gridDim.y << 4;
    const int row_base = ct * 16;
    const int col      = mt * 16 + (lane & 15);
    const int hi       = lane >> 4;
    const int arow     = row_base + (lane & 15);

    __shared__ float lsum[16], lsq[16];
    if (threadIdx.x < 16) { lsum[threadIdx.x] = 0.0f; lsq[threadIdx.x] = 0.0f; }
    __syncthreads();

    const _Float16* xcol = X + (size_t)col * CSTORE;

    v8f acc = {};
#pragma unroll
    for (int ks = 0; ks < KTOT; ks += 32) {
        const int kb0 = ks + hi * 8;        // first 8-half chunk base
        const int kb1 = ks + 16 + hi * 8;   // second 8-half chunk base

        // ---- A fragment (weights, f32 -> f16) ----
        v16h a;
        if constexpr (KTOT == 64) {
            const float4 wa = *(const float4*)(W + arow * 64 + kb0);
            const float4 wb = *(const float4*)(W + arow * 64 + kb0 + 4);
            const float4 wc = *(const float4*)(W + arow * 64 + kb1);
            const float4 wd = *(const float4*)(W + arow * 64 + kb1 + 4);
            a[0]  = (_Float16)wa.x; a[1]  = (_Float16)wa.y; a[2]  = (_Float16)wa.z; a[3]  = (_Float16)wa.w;
            a[4]  = (_Float16)wb.x; a[5]  = (_Float16)wb.y; a[6]  = (_Float16)wb.z; a[7]  = (_Float16)wb.w;
            a[8]  = (_Float16)wc.x; a[9]  = (_Float16)wc.y; a[10] = (_Float16)wc.z; a[11] = (_Float16)wc.w;
            a[12] = (_Float16)wd.x; a[13] = (_Float16)wd.y; a[14] = (_Float16)wd.z; a[15] = (_Float16)wd.w;
        } else {
#pragma unroll
            for (int h = 0; h < 8; ++h) {
                a[h]     = (kb0 + h < cin) ? (_Float16)W[arow * cin + kb0 + h] : (_Float16)0.0f;
                a[h + 8] = (kb1 + h < cin) ? (_Float16)W[arow * cin + kb1 + h] : (_Float16)0.0f;
            }
        }

        // ---- B fragment (activations, two 16B vector loads) ----
        v8h c0 = {}, c1 = {};
        if (kb0 < CSTORE) c0 = *(const v8h*)(xcol + kb0);
        if (kb1 < CSTORE) c1 = *(const v8h*)(xcol + kb1);
        v16h b;
#pragma unroll
        for (int h = 0; h < 8; ++h) {
            if constexpr (AFFINE) {
                float x0 = fmaxf(fmaf((float)c0[h], aff_a[kb0 + h], aff_c[kb0 + h]), 0.0f);
                float x1 = fmaxf(fmaf((float)c1[h], aff_a[kb1 + h], aff_c[kb1 + h]), 0.0f);
                b[h]     = (_Float16)x0;
                b[h + 8] = (_Float16)x1;
            } else {
                b[h]     = c0[h];
                b[h + 8] = c1[h];
            }
        }

        acc = __builtin_amdgcn_wmma_f32_16x16x32_f16(
            /*neg_a=*/false, a, /*neg_b=*/false, b,
            /*c_mod=*/(short)0, acc, /*reuse_a=*/false, /*reuse_b=*/false);
    }

    // ---- store D tile: 8 contiguous halfs -> one 16-byte store per lane ----
    v8h d;
#pragma unroll
    for (int r = 0; r < 8; ++r) d[r] = (_Float16)acc[r];
    *(v8h*)(Y + (size_t)col * cout + row_base + hi * 8) = d;

    // ---- fused BN stats: column-reduce within each 16-lane half ----
#pragma unroll
    for (int r = 0; r < 8; ++r) {
        float s = acc[r];
        float q = acc[r] * acc[r];
#pragma unroll
        for (int m = 1; m < 16; m <<= 1) {
            s += __shfl_xor(s, m);
            q += __shfl_xor(q, m);
        }
        if ((lane & 15) == 0) {
            atomicAdd(&lsum[hi * 8 + r], s);
            atomicAdd(&lsq[hi * 8 + r], q);
        }
    }
    __syncthreads();
    if (threadIdx.x < 16) {
        atomicAdd(&sum[row_base + threadIdx.x],   lsum[threadIdx.x]);
        atomicAdd(&sumsq[row_base + threadIdx.x], lsq[threadIdx.x]);
    }
}

// ---------------------------------------------------------------------------
// 5) BN fused affine params: a = g*rsqrt(var+eps), c = beta - mean*a.
// ---------------------------------------------------------------------------
__global__ void bn_params_kernel(const float* __restrict__ sum,
                                 const float* __restrict__ sumsq,
                                 const float* __restrict__ g,
                                 const float* __restrict__ beta,
                                 float* __restrict__ a, float* __restrict__ c,
                                 int C) {
    const int i = blockIdx.x * blockDim.x + threadIdx.x;
    if (i >= C) return;
    const float invM = 1.0f / (float)MTOT;
    const float mean = sum[i] * invM;
    const float var  = sumsq[i] * invM - mean * mean;
    const float sc   = g[i] * rsqrtf(var + EPS_);
    a[i] = sc;
    c[i] = beta[i] - mean * sc;
}

// ---------------------------------------------------------------------------
// 6) Maxpool over NS: one wave per (b,s,channel-group-of-8). lane = ns.
//    16-byte vector load per lane, shfl_xor max tree across the 32 samples.
//    Y2 layout [M][128]; out (B,128,S).
// ---------------------------------------------------------------------------
__global__ void maxpool_kernel(const _Float16* __restrict__ Y2,
                               const float* __restrict__ a,
                               const float* __restrict__ cc,
                               float* __restrict__ out) {
    const int w    = (int)((blockIdx.x * blockDim.x + threadIdx.x) >> 5);
    const int lane = threadIdx.x & 31;
    const int bs = w >> 4;          // (b*S+s)
    const int cg = w & 15;          // channel group (8 channels)
    const int b  = bs / S_;
    const int s  = bs % S_;

    const v8h v = *(const v8h*)(Y2 + ((size_t)bs * NS_ + lane) * 128 + cg * 8);
    float f[8];
#pragma unroll
    for (int j = 0; j < 8; ++j) {
        const int c = cg * 8 + j;
        f[j] = fmaxf(fmaf((float)v[j], a[c], cc[c]), 0.0f);
    }
#pragma unroll
    for (int j = 0; j < 8; ++j) {
#pragma unroll
        for (int m = 1; m < 32; m <<= 1) f[j] = fmaxf(f[j], __shfl_xor(f[j], m));
    }
    if (lane == 0) {
#pragma unroll
        for (int j = 0; j < 8; ++j)
            out[(size_t)b * 128 * S_ + (size_t)(cg * 8 + j) * S_ + s] = f[j];
    }
}

// new_xyz scratch (B,S,3) -> output (B,3,S)
__global__ void write_newxyz_kernel(const float* __restrict__ nx,
                                    float* __restrict__ out) {
    const int i = blockIdx.x * blockDim.x + threadIdx.x;
    if (i >= B_ * 3 * S_) return;
    const int s = i % S_;
    const int d = (i / S_) % 3;
    const int b = i / (3 * S_);
    out[i] = nx[((size_t)b * S_ + s) * 3 + d];
}

__global__ void zero_kernel(float* __restrict__ p, int n) {
    const int i = blockIdx.x * blockDim.x + threadIdx.x;
    if (i < n) p[i] = 0.0f;
}

// ---------------------------------------------------------------------------
// Host launch
// ---------------------------------------------------------------------------
extern "C" void kernel_launch(void* const* d_in, const int* in_sizes, int n_in,
                              void* d_out, int out_size, void* d_ws, size_t ws_size,
                              hipStream_t stream) {
    const float* xyz    = (const float*)d_in[0];
    const float* points = (const float*)d_in[1];
    const float* w0  = (const float*)d_in[2];
    const float* g0  = (const float*)d_in[4];
    const float* be0 = (const float*)d_in[5];
    const float* w1  = (const float*)d_in[6];
    const float* g1  = (const float*)d_in[8];
    const float* be1 = (const float*)d_in[9];
    const float* w2  = (const float*)d_in[10];
    const float* g2  = (const float*)d_in[12];
    const float* be2 = (const float*)d_in[13];

    char* ws = (char*)d_ws;
    size_t off = 0;
    auto take = [&](size_t bytes) -> char* {
        char* p = ws + off;
        off = (off + bytes + 255) & ~(size_t)255;
        return p;
    };
    int*      fps_idx = (int*)take((size_t)B_ * S_ * 4);
    float*    nxyz    = (float*)take((size_t)B_ * S_ * 3 * 4);
    int*      grp     = (int*)take((size_t)MTOT * 4);
    float*    sums    = (float*)take(512 * 4);   // [s0|q0|s1|q1|s2(128)|q2(128)]
    float*    bn      = (float*)take(512 * 4);   // [a0|c0|a1|c1|a2(128)|c2(128)]
    _Float16* bufA    = (_Float16*)take((size_t)64  * MTOT * 2); // X0 then y1
    _Float16* bufB    = (_Float16*)take((size_t)128 * MTOT * 2); // y0 then y2

    float* out_xyz = (float*)d_out;
    float* out_pts = (float*)d_out + (size_t)B_ * 3 * S_;

    zero_kernel<<<2, 256, 0, stream>>>(sums, 512);

    fps_kernel<<<B_, 256, 0, stream>>>(xyz, fps_idx, nxyz);
    write_newxyz_kernel<<<(B_ * 3 * S_ + 255) / 256, 256, 0, stream>>>(nxyz, out_xyz);

    query_ball_kernel<<<(B_ * S_) / 8, 256, 0, stream>>>(xyz, nxyz, grp);
    build_feat_kernel<<<MTOT / 256, 256, 0, stream>>>(xyz, points, nxyz, grp, bufA);

    const dim3 gemm_blk(256);
    // Layer 0: 64x6 (pad K->32, 8 stored channels), X0 raw -> y0 [M][64] (bufB)
    gemm_wmma_kernel<32, 8, false><<<dim3(MTOT / 16 / 8, 4), gemm_blk, 0, stream>>>(
        w0, 6, bufA, nullptr, nullptr, bufB, sums + 0, sums + 64);
    bn_params_kernel<<<1, 64, 0, stream>>>(sums + 0, sums + 64, g0, be0, bn + 0, bn + 64, 64);

    // Layer 1: 64x64, BN0 affine+ReLU fused into loads -> y1 [M][64] (bufA, X0 dead)
    gemm_wmma_kernel<64, 64, true><<<dim3(MTOT / 16 / 8, 4), gemm_blk, 0, stream>>>(
        w1, 64, bufB, bn + 0, bn + 64, bufA, sums + 128, sums + 192);
    bn_params_kernel<<<1, 64, 0, stream>>>(sums + 128, sums + 192, g1, be1, bn + 128, bn + 192, 64);

    // Layer 2: 128x64, BN1 affine+ReLU fused -> y2 [M][128] (bufB, y0 dead)
    gemm_wmma_kernel<64, 64, true><<<dim3(MTOT / 16 / 8, 8), gemm_blk, 0, stream>>>(
        w2, 64, bufA, bn + 128, bn + 192, bufB, sums + 256, sums + 384);
    bn_params_kernel<<<1, 128, 0, stream>>>(sums + 256, sums + 384, g2, be2, bn + 256, bn + 384, 128);

    // BN2 affine + ReLU + max over NS -> new_points (B,128,S)
    maxpool_kernel<<<(B_ * S_ * 16) / 8, 256, 0, stream>>>(bufB, bn + 256, bn + 384, out_pts);
}
